// LearnableTensorSquare_36636071035616
// MI455X (gfx1250) — matrix operand order repaired and checked by reading, not compiled, not run on hardware
//
#include <hip/hip_runtime.h>
#include <stdint.h>

// ---------- problem constants ----------
#define DIM_IN   64
#define DIM_TS   256
#define DIM_OUT  64
#define NNZ      32768
#define Z_TOTAL  65536
#define KK       (DIM_IN * DIM_IN)   // 4096
#define CHUNK    512                 // kk-dim chunk staged in LDS (double-buffered)
#define NCHUNK   (KK / CHUNK)        // 8
// Row layout in LDS: 512 bf16 + 8 bf16 pad at row end
// => row stride 520 bf16 = 1040 B = 260 DW => 4-bank rotation per row (no conflicts).
#define SB_STRIDE 520
#define SF_STRIDE 72                 // padded F row stride (bf16): 144B

typedef __bf16 v16bf __attribute__((ext_vector_type(16)));
typedef float  v8f   __attribute__((ext_vector_type(8)));
typedef unsigned int u32x4 __attribute__((ext_vector_type(4)));
typedef int          i32x4 __attribute__((ext_vector_type(4)));
typedef int          i32x8 __attribute__((ext_vector_type(8)));

struct BU8 { uint32_t x[8]; };       // 32-byte carrier for bit_cast to v16bf

static __device__ __forceinline__ unsigned short f32_to_bf16(float f) {
  unsigned u = __float_as_uint(f);
  u += 0x7FFFu + ((u >> 16) & 1u);   // round-to-nearest-even
  return (unsigned short)(u >> 16);
}

// ---------- prologue kernels ----------
__global__ void zero_f32(float* __restrict__ p, int n) {
  int i = blockIdx.x * blockDim.x + threadIdx.x;
  if (i < n) p[i] = 0.0f;
}

// kernel_flat[k, col] += W[k, row] * val   (COO scatter of T, fused with W mix)
__global__ void scatter_WT(const float* __restrict__ W,
                           const float* __restrict__ vals,
                           const int*   __restrict__ rows,
                           const int*   __restrict__ cols,
                           float*       __restrict__ kflat) {
  int e = blockIdx.x * blockDim.x + threadIdx.x;
  if (e >= NNZ) return;
  const float v = vals[e];
  const int   r = rows[e];
  const int   c = cols[e];
  #pragma unroll 4
  for (int k = 0; k < DIM_OUT; ++k) {
    atomicAdd(&kflat[(size_t)k * KK + c], W[(size_t)k * DIM_TS + r] * v);
  }
}

__global__ void cvt_bf16(const float* __restrict__ src,
                         unsigned short* __restrict__ dst, int n) {
  int i = blockIdx.x * blockDim.x + threadIdx.x;
  if (i < n) dst[i] = f32_to_bf16(src[i]);
}

// ---------- main fused kernel: out = FF @ kernel_flat^T via WMMA bf16 ----------
// 128 threads = 4 waves. Wave w owns TWO 16-row z-tiles (32 rows) and all 64 k's:
// each B fragment read from LDS feeds two WMMAs (halves DS traffic per MAC).
// kernel_flat^T streamed through double-buffered LDS chunks via the TDM,
// with the next chunk's DMA overlapped with current-chunk compute.
__global__ __launch_bounds__(128)
void lts_main(const float* __restrict__ features,
              const unsigned short* __restrict__ kbf,   // kernel_flat bf16 [64, 4096] row-major
              float* __restrict__ out) {
  __shared__ __align__(16) unsigned short sB[2][DIM_OUT * SB_STRIDE]; // 2 x 65 KB
  __shared__ __align__(16) unsigned short sF[4 * 32 * SF_STRIDE];     // 18 KB

  const int tid  = threadIdx.x;
  const int w    = tid >> 5;        // wave 0..3
  const int lane = tid & 31;
  const int n    = lane & 15;       // WMMA column / A row m
  const int half = lane >> 4;       // lane-half selects K sub-range
  const int ztw  = blockIdx.x * 128 + w * 32;   // wave's 32 z rows

  // Stage this wave's 32x64 f32 feature rows into LDS as bf16.
  {
    unsigned short* f = &sF[w * 32 * SF_STRIDE];
    #pragma unroll
    for (int it = 0; it < 16; ++it) {
      int idx = lane + it * 32;            // 0..511 -> (row, 4-float segment)
      int row = idx >> 4;                  // 0..31
      int seg = idx & 15;
      const float4 v = *(const float4*)&features[(size_t)(ztw + row) * DIM_IN + seg * 4];
      unsigned short* d = &f[row * SF_STRIDE + seg * 4];
      d[0] = f32_to_bf16(v.x); d[1] = f32_to_bf16(v.y);
      d[2] = f32_to_bf16(v.z); d[3] = f32_to_bf16(v.w);
    }
  }

  // Lane's A rows (row m == n) for the two z-tiles.
  const unsigned short* fw0 = &sF[(w * 32 + n) * SF_STRIDE];
  const unsigned short* fw1 = fw0 + 16 * SF_STRIDE;

  // fj operands are i-independent: preload once per z-tile (j 0..31 / 32..63 variants).
  v16bf fj00, fj01, fj10, fj11;
  {
    BU8 t;
    *(uint4*)&t.x[0] = *(const uint4*)&fw0[ 0 + half * 8];
    *(uint4*)&t.x[4] = *(const uint4*)&fw0[16 + half * 8];
    fj00 = __builtin_bit_cast(v16bf, t);
    *(uint4*)&t.x[0] = *(const uint4*)&fw0[32 + half * 8];
    *(uint4*)&t.x[4] = *(const uint4*)&fw0[48 + half * 8];
    fj01 = __builtin_bit_cast(v16bf, t);
    *(uint4*)&t.x[0] = *(const uint4*)&fw1[ 0 + half * 8];
    *(uint4*)&t.x[4] = *(const uint4*)&fw1[16 + half * 8];
    fj10 = __builtin_bit_cast(v16bf, t);
    *(uint4*)&t.x[0] = *(const uint4*)&fw1[32 + half * 8];
    *(uint4*)&t.x[4] = *(const uint4*)&fw1[48 + half * 8];
    fj11 = __builtin_bit_cast(v16bf, t);
  }

#define HAVE_TDM (__has_builtin(__builtin_amdgcn_tensor_load_to_lds) && \
                  __has_builtin(__builtin_amdgcn_s_wait_tensorcnt))

#if HAVE_TDM
  // Issue one 2D TDM: kernel_flat[:, c*512 .. +512) -> sB[buf], 16B pad per row.
  #define TDM_ISSUE(BUF, C)                                                         \
  {                                                                                 \
    const uint64_t ga = (uint64_t)(uintptr_t)(kbf + (size_t)(C) * CHUNK);           \
    const u32x4 g0 = { 1u,                                                          \
                       (unsigned)(uintptr_t)&sB[(BUF)][0],                          \
                       (unsigned)ga,                                                \
                       (unsigned)((ga >> 32) & 0x1FFFFFFu) | (2u << 30) };          \
    const i32x8 g1 = { (int)((1u << 16) | (1u << 20) | (7u << 22) | (3u << 25)),    \
                       (int)(4096u << 16),   /* tensor_dim0 = 4096 */               \
                       (int)(64u   << 16),   /* tensor_dim1 = 64   */               \
                       (int)(512u  << 16),   /* tile_dim0   = 512  */               \
                       (int)64,              /* tile_dim1   = 64   */               \
                       (int)4096,            /* tensor_dim0_stride */               \
                       0, 0 };                                                      \
    const i32x4 gz = { 0, 0, 0, 0 };                                                \
    TDM_CALL(g0, g1, gz);                                                           \
  }
  #if __clang_major__ >= 23
    #define TDM_CALL(G0, G1, GZ)                                                    \
      { const i32x8 gz8 = { 0,0,0,0,0,0,0,0 };                                      \
        __builtin_amdgcn_tensor_load_to_lds((G0), (G1), (GZ), (GZ), gz8, 0); }
  #else
    #define TDM_CALL(G0, G1, GZ)                                                    \
        __builtin_amdgcn_tensor_load_to_lds((G0), (G1), (GZ), (GZ), 0)
  #endif
#else
  // Fallback: cooperative vector copy with the same padded layout.
  #define TDM_ISSUE_ALL(BUF, C)                                                     \
  {                                                                                 \
    _Pragma("unroll")                                                               \
    for (int it = 0; it < 32; ++it) {                                               \
      int linear = tid + it * 128;      /* 4096 x 16-byte chunks */                 \
      int row  = linear >> 6;                                                       \
      int colc = linear & 63;                                                       \
      const uint4 v = *(const uint4*)&kbf[(size_t)row * KK + (C) * CHUNK + colc*8]; \
      *(uint4*)&sB[(BUF)][row * SB_STRIDE + colc * 8] = v;                          \
    }                                                                               \
  }
#endif

  // ---- prologue: load chunk 0 ----
#if HAVE_TDM
  if (w == 0) {
    TDM_ISSUE(0, 0)
    __builtin_amdgcn_s_wait_tensorcnt(0);
  }
#else
  TDM_ISSUE_ALL(0, 0)
#endif
  __syncthreads();

  v8f a00 = {}, a01 = {}, a02 = {}, a03 = {};   // z-tile 0, N-tiles 0..3
  v8f a10 = {}, a11 = {}, a12 = {}, a13 = {};   // z-tile 1

  for (int c = 0; c < NCHUNK; ++c) {
    const int cur = c & 1;

    // Overlap: start DMA of the next chunk into the idle buffer.
    if (c + 1 < NCHUNK) {
#if HAVE_TDM
      if (w == 0) TDM_ISSUE(1 - cur, c + 1)
#else
      TDM_ISSUE_ALL(1 - cur, c + 1)
#endif
    }

    // per-chunk f[m, i] scalars: 8 values per z-tile, packed 2/VGPR
    uint32_t fiP0[4], fiP1[4];
    *(uint4*)&fiP0[0] = *(const uint4*)&fw0[c * 8];
    *(uint4*)&fiP1[0] = *(const uint4*)&fw1[c * 8];

    const unsigned short* sbc = &sB[cur][0];

    #pragma unroll
    for (int s = 0; s < 16; ++s) {           // 16 K-steps of 32
      const int iw   = s >> 2;               // packed word
      const int hsel = (s >> 1) & 1;         // which bf16 half
      const int jbs  = s & 1;                // jb = 0 / 32

      const unsigned short fiu0 =
          (unsigned short)(hsel ? (fiP0[iw] >> 16) : (fiP0[iw] & 0xFFFFu));
      const unsigned short fiu1 =
          (unsigned short)(hsel ? (fiP1[iw] >> 16) : (fiP1[iw] & 0xFFFFu));
      const __bf16 fib0 = __builtin_bit_cast(__bf16, fiu0);
      const __bf16 fib1 = __builtin_bit_cast(__bf16, fiu1);
      v16bf fs0, fs1;
      #pragma unroll
      for (int e = 0; e < 16; ++e) { fs0[e] = fib0; fs1[e] = fib1; }

      const v16bf A0 = fs0 * (jbs ? fj01 : fj00);   // 8x v_pk_mul_bf16 each
      const v16bf A1 = fs1 * (jbs ? fj11 : fj10);

      const int bco = s * 32 + half * 8;            // K-range within sB row

      #define DO_NT(NT, ACC0, ACC1)                                                 \
      {                                                                             \
        BU8 bu;                                                                     \
        const unsigned short* bp = &sbc[((NT) * 16 + n) * SB_STRIDE + bco];         \
        *(uint4*)&bu.x[0] = *(const uint4*)&bp[0];                                  \
        *(uint4*)&bu.x[4] = *(const uint4*)&bp[16];                                 \
        const v16bf B = __builtin_bit_cast(v16bf, bu);                              \
        ACC0 = __builtin_amdgcn_wmma_f32_16x16x32_bf16(                             \
                   false, A0, false, B, (short)0, ACC0, false, false);              \
        ACC1 = __builtin_amdgcn_wmma_f32_16x16x32_bf16(                             \
                   false, A1, false, B, (short)0, ACC1, false, false);              \
      }
      DO_NT(0, a00, a10)
      DO_NT(1, a01, a11)
      DO_NT(2, a02, a12)
      DO_NT(3, a03, a13)
      #undef DO_NT
    }

    if (c + 1 < NCHUNK) {
#if HAVE_TDM
      if (w == 0) __builtin_amdgcn_s_wait_tensorcnt(0);
#endif
      __syncthreads();   // next buffer ready for everyone; current buffer idle
    }
  }

  // D layout: element r -> (z = base + half*8 + r, k = nt*16 + n); coalesced 64B runs.
  {
    float* o0 = out + (size_t)(ztw + half * 8) * DIM_OUT + n;
    float* o1 = o0 + (size_t)16 * DIM_OUT;
    #pragma unroll
    for (int r = 0; r < 8; ++r) {
      float* p0 = o0 + (size_t)r * DIM_OUT;
      float* p1 = o1 + (size_t)r * DIM_OUT;
      p0[0] = a00[r]; p0[16] = a01[r]; p0[32] = a02[r]; p0[48] = a03[r];
      p1[0] = a10[r]; p1[16] = a11[r]; p1[32] = a12[r]; p1[48] = a13[r];
    }
  }
}

// ---------- launch ----------
extern "C" void kernel_launch(void* const* d_in, const int* in_sizes, int n_in,
                              void* d_out, int out_size, void* d_ws, size_t ws_size,
                              hipStream_t stream) {
  const float* features = (const float*)d_in[0];
  const float* W        = (const float*)d_in[1];
  const float* T_values = (const float*)d_in[2];
  const int*   T_row    = (const int*)d_in[3];
  const int*   T_col    = (const int*)d_in[4];
  float* out = (float*)d_out;

  float*          kflat = (float*)d_ws;                                  // 1 MiB
  unsigned short* kbf   = (unsigned short*)((char*)d_ws +
                            (size_t)DIM_OUT * KK * sizeof(float));       // +512 KiB

  const int NK = DIM_OUT * KK;  // 262144

  hipLaunchKernelGGL(zero_f32,   dim3((NK + 255) / 256),  dim3(256), 0, stream, kflat, NK);
  hipLaunchKernelGGL(scatter_WT, dim3((NNZ + 255) / 256), dim3(256), 0, stream,
                     W, T_values, T_row, T_col, kflat);
  hipLaunchKernelGGL(cvt_bf16,   dim3((NK + 255) / 256),  dim3(256), 0, stream, kflat, kbf, NK);
  hipLaunchKernelGGL(lts_main,   dim3(Z_TOTAL / 128),     dim3(128), 0, stream,
                     features, kbf, out);
}
